// AttentionLayer_40656160424479
// MI455X (gfx1250) — matrix operand order repaired and checked by reading, not compiled, not run on hardware
//
#include <hip/hip_runtime.h>
#include <math.h>

typedef __attribute__((ext_vector_type(16))) _Float16 v16h;
typedef __attribute__((ext_vector_type(8)))  _Float16 v8h;
typedef __attribute__((ext_vector_type(8)))  float    v8f;
typedef int v4i __attribute__((vector_size(16)));   // matches builtin's param type

constexpr int kSL = 2048;
constexpr int kBS = 4;
constexpr int kD  = 512;
constexpr int kH  = 8;
constexpr int kNH = 64;
constexpr float kScale = 0.125f; // 1/sqrt(64)

// ---- optional CDNA5 async-to-LDS path (probe via __has_builtin) -----------
#if defined(__HIP_DEVICE_COMPILE__) && defined(__has_builtin)
#  if __has_builtin(__builtin_amdgcn_global_load_async_to_lds_b128) && \
      __has_builtin(__builtin_amdgcn_s_wait_asynccnt)
#    define HAVE_ASYNC 1
#  endif
#endif
#ifndef HAVE_ASYNC
#define HAVE_ASYNC 0
#endif

#if HAVE_ASYNC
__device__ __forceinline__ void async_copy_b128(const _Float16* g, _Float16* l) {
    __builtin_amdgcn_global_load_async_to_lds_b128(
        (__attribute__((address_space(1))) v4i*)(v4i*)(void*)g,
        (__attribute__((address_space(3))) v4i*)(v4i*)(void*)l,
        0, 0);
}
#endif

// ---------------------------------------------------------------------------
// A-fragment loader for v_wmma_f32_16x16x32_f16.
// Lane L holds row M = L&15; halves i map to K = i + (L<16?0:16).
// Caller passes p = rowbase + kb + (L<16?0:8).
// ---------------------------------------------------------------------------
__device__ __forceinline__ v16h load_a16(const _Float16* p) {
    p += ((threadIdx.x >> 4) & 1) * 8;
    v8h lo = *(const v8h*)(p);        // K = c .. c+7
    v8h hi = *(const v8h*)(p + 8);    // K = c+8 .. c+15
    v16h a;
#pragma unroll
    for (int i = 0; i < 8; ++i) { a[i] = lo[i]; a[i + 8] = hi[i]; }
    return a;
}

// ---------------------------------------------------------------------------
__global__ void cvt_f16_kernel(const float* __restrict__ in,
                               _Float16* __restrict__ out, int n) {
    int i = blockIdx.x * blockDim.x + threadIdx.x;
    if (i < n) out[i] = (_Float16)in[i];
}

// W[k][n] (f32, 512x512) -> Wt[n][k] (f16) so B-fragments load contiguously.
__global__ void transpose_w_kernel(const float* __restrict__ W,
                                   _Float16* __restrict__ Wt) {
    int n = blockIdx.x * 16 + threadIdx.x;
    int k = blockIdx.y * 16 + threadIdx.y;
    Wt[(size_t)n * kD + k] = (_Float16)W[(size_t)k * kD + n];
}

// ---------------------------------------------------------------------------
// GEMM: Y[M,512] = X[M,512](f16) @ Wt[n][k](f16) + bias, via WMMA.
// One wave computes a 32x64 tile (2x4 C fragments; B reused across 2 A rows).
// mode 0: f16 row-major out   (Q, K projections)
// mode 1: f16 V-transposed    Vt[((b*H+h)*64+nh)*SL + s]
// mode 2: f32 row-major out   (final Wo projection -> d_out)
// ---------------------------------------------------------------------------
__global__ void __launch_bounds__(256)
gemm512_wmma(const _Float16* __restrict__ X,
             const _Float16* __restrict__ Wt,
             const float* __restrict__ bias,
             void* __restrict__ out, int M, int mode) {
    const int lane = threadIdx.x;          // 0..31
    const int wave = threadIdx.y;          // 0..7
    const int m0   = (blockIdx.y * 8 + wave) * 32;
    const int n0   = blockIdx.x * 64;
    if (m0 >= M) return;

    const int lh  = (lane < 16) ? 0 : 1;
    const int col = lane & 15;

    v8f acc[2][4] = { { {}, {}, {}, {} }, { {}, {}, {}, {} } };

    const _Float16* xr0 = X + (size_t)(m0 + col) * kD + lh * 8;
    const _Float16* xr1 = xr0 + (size_t)16 * kD;
#pragma unroll 2
    for (int kb = 0; kb < kD; kb += 32) {
        v16h a0 = load_a16(xr0 + kb);
        v16h a1 = load_a16(xr1 + kb);
#pragma unroll
        for (int c = 0; c < 4; ++c) {
            const int n = n0 + c * 16 + col;
            v16h bf = *(const v16h*)(Wt + (size_t)n * kD + kb + lh * 16);
            acc[0][c] = __builtin_amdgcn_wmma_f32_16x16x32_f16(
                false, a0, false, bf, (short)0, acc[0][c], false, false);
            acc[1][c] = __builtin_amdgcn_wmma_f32_16x16x32_f16(
                false, a1, false, bf, (short)0, acc[1][c], false, false);
        }
    }

#pragma unroll
    for (int hm = 0; hm < 2; ++hm) {
        const int rbase = m0 + hm * 16 + lh * 8;
#pragma unroll
        for (int c = 0; c < 4; ++c) {
            const int n  = n0 + c * 16 + col;
            const float bv = bias[n];
#pragma unroll
            for (int r = 0; r < 8; ++r) {
                const int row = rbase + r;
                const float val = acc[hm][c][r] + bv;
                if (mode == 0) {
                    ((_Float16*)out)[(size_t)row * kD + n] = (_Float16)val;
                } else if (mode == 1) {
                    const int s = row >> 2, b = row & 3;   // row = s*BS + b
                    const int h = n >> 6, nh = n & 63;     // n   = h*64 + nh
                    ((_Float16*)out)[(size_t)((b * kH + h) * kNH + nh) * kSL + s] =
                        (_Float16)val;
                } else {
                    ((float*)out)[(size_t)row * kD + n] = val;
                }
            }
        }
    }
}

// ---------------------------------------------------------------------------
// Causal flash attention, one wave per (16-query tile, b, h).
// K/V tiles double-buffered in LDS; next tile fetched with async-to-LDS
// (ASYNCcnt) while WMMAs run on the current tile.
// Q,K: f16 [M=8192, 512] row-major.  Vt: f16 [(b,h,nh), s] transposed.
// O:   f16 [M, 512] row-major.
// ---------------------------------------------------------------------------
__global__ void __launch_bounds__(32)
attn_fa_wmma(const _Float16* __restrict__ Qh,
             const _Float16* __restrict__ Kh,
             const _Float16* __restrict__ Vt,
             _Float16* __restrict__ Oh) {
    __shared__ _Float16 sK[2][32 * kNH];   // [key_local][nh]
    __shared__ _Float16 sV[2][kNH * 32];   // [nh][key_local]
    __shared__ float    sS[16 * 32];
    __shared__ _Float16 sP[16 * 32];
    __shared__ float    sBr[16];

    const int lane = threadIdx.x;
    const int q0   = blockIdx.x * 16;
    const int b    = blockIdx.y >> 3;
    const int h    = blockIdx.y & 7;
    const int lh   = (lane < 16) ? 0 : 1;
    const int col  = lane & 15;

    const size_t kvoff = (size_t)b * kD + h * kNH;          // within a key row
    const size_t vbase = (size_t)(b * kH + h) * kNH * kSL;  // Vt (b,h) slab

    // stage one 32-key K/V tile pair (8 KB) into LDS buffer `buf`
    auto stage = [&](int buf, int jb) {
#pragma unroll
        for (int i = 0; i < 8; ++i) {
            const int c   = i * 32 + lane;        // 256 x 16B chunks
            const int key = c >> 3, ko = (c & 7) * 8;
            const _Float16* gk = Kh + (size_t)(jb + key) * (kBS * kD) + kvoff + ko;
            _Float16*       lk = &sK[buf][key * kNH + ko];
            const int nh = c >> 2, vo = (c & 3) * 8;
            const _Float16* gv = Vt + vbase + (size_t)nh * kSL + jb + vo;
            _Float16*       lv = &sV[buf][nh * 32 + vo];
#if HAVE_ASYNC
            async_copy_b128(gk, lk);
            async_copy_b128(gv, lv);
#else
            *(v8h*)lk = *(const v8h*)gk;
            *(v8h*)lv = *(const v8h*)gv;
#endif
        }
    };

    // Q fragments (rows q0..q0+15, head slice), K-dim blocks 0..31 / 32..63.
    const _Float16* qrow =
        Qh + ((size_t)(q0 + col) * kBS + b) * kD + h * kNH + lh * 8;
    const v16h qa0 = load_a16(qrow);
    const v16h qa1 = load_a16(qrow + 32);

    v8f   o[4] = { {}, {}, {}, {} };
    float mrow = -INFINITY, lrow = 0.f;   // meaningful in lanes 0..15

    const int jb_end = (q0 + 15) & ~31;
    stage(0, 0);
    for (int jb = 0; jb <= jb_end; jb += 32) {
        const int buf = (jb >> 5) & 1;
#if HAVE_ASYNC
        __builtin_amdgcn_s_wait_asynccnt(0);   // current tile resident
#else
        __syncthreads();
#endif
        if (jb + 32 <= jb_end) stage(buf ^ 1, jb + 32);  // overlap next fetch

        const _Float16* sKb = sK[buf];
        const _Float16* sVb = sV[buf];

        // ----- S = Q K^T for 32 keys, as two 16x16 fragments -----
#pragma unroll
        for (int g = 0; g < 2; ++g) {
            const _Float16* kr = sKb + (g * 16 + col) * kNH + lh * 16;
            v16h k0 = *(const v16h*)(kr);        // nh 0..31 slice
            v16h k1 = *(const v16h*)(kr + 32);   // nh 32..63 slice
            v8f s = {};
            s = __builtin_amdgcn_wmma_f32_16x16x32_f16(
                false, qa0, false, k0, (short)0, s, false, false);
            s = __builtin_amdgcn_wmma_f32_16x16x32_f16(
                false, qa1, false, k1, (short)0, s, false, false);
#pragma unroll
            for (int r = 0; r < 8; ++r)
                sS[(r + lh * 8) * 32 + g * 16 + col] = s[r];
        }
        __syncthreads();

        // ----- streaming softmax: one row per lane (lanes 0..15) -----
        if (lane < 16) {
            const int qi = q0 + lane;
            float sv[32];
            float mx = mrow;
#pragma unroll
            for (int c = 0; c < 32; ++c) {
                const float x = sS[lane * 32 + c] * kScale;
                sv[c] = (jb + c <= qi) ? x : -1e30f;
                mx = fmaxf(mx, sv[c]);
            }
            const float alpha = __expf(mrow - mx);
            float ls = 0.f;
#pragma unroll
            for (int c = 0; c < 32; ++c) {
                const float e = __expf(sv[c] - mx);
                ls += e;
                sP[lane * 32 + c] = (_Float16)e;
            }
            lrow = lrow * alpha + ls;
            mrow = mx;
            sBr[lane] = alpha;
        }
        __syncthreads();

        // rescale O accumulators by per-row alpha
#pragma unroll
        for (int r = 0; r < 8; ++r) {
            const float a = sBr[r + lh * 8];
#pragma unroll
            for (int c = 0; c < 4; ++c) o[c][r] *= a;
        }

        // P as A-fragment (16 rows x 32 keys)
        const v16h pa = load_a16(sP + col * 32 + lh * 8);

        // ----- O += P @ V  (V tile 32 keys x 64 nh, 4 B fragments) -----
#pragma unroll
        for (int c = 0; c < 4; ++c) {
            const _Float16* vr = sVb + (c * 16 + col) * 32 + lh * 16;
            v16h vb = *(const v16h*)vr;
            o[c] = __builtin_amdgcn_wmma_f32_16x16x32_f16(
                false, pa, false, vb, (short)0, o[c], false, false);
        }
        __syncthreads();
    }

    // ----- normalize by l and write O (f16 row-major) -----
    if (lane < 16) sBr[lane] = lrow;
    __syncthreads();
#pragma unroll
    for (int r = 0; r < 8; ++r) {
        const float inv = 1.f / sBr[r + lh * 8];
        const int row = q0 + r + lh * 8;
        _Float16* orow = Oh + ((size_t)row * kBS + b) * kD + h * kNH;
#pragma unroll
        for (int c = 0; c < 4; ++c)
            orow[c * 16 + col] = (_Float16)(o[c][r] * inv);
    }
}

// ---------------------------------------------------------------------------
extern "C" void kernel_launch(void* const* d_in, const int* in_sizes, int n_in,
                              void* d_out, int out_size, void* d_ws, size_t ws_size,
                              hipStream_t stream) {
    const float* x  = (const float*)d_in[0];
    const float* kx = (const float*)d_in[1];
    const float* vx = (const float*)d_in[2];
    const float* Wq = (const float*)d_in[3];
    const float* bq = (const float*)d_in[4];
    const float* Wk = (const float*)d_in[5];
    const float* bk = (const float*)d_in[6];
    const float* Wv = (const float*)d_in[7];
    const float* bv = (const float*)d_in[8];
    const float* Wo = (const float*)d_in[9];
    const float* bo = (const float*)d_in[10];
    float* out = (float*)d_out;

    const size_t NE = (size_t)kSL * kBS * kD;  // 4,194,304 elems
    const size_t WE = (size_t)kD * kD;         //   262,144 elems

    _Float16* p   = (_Float16*)d_ws;
    _Float16* xh  = p; p += NE;
    _Float16* kh  = p; p += NE;
    _Float16* vh  = p; p += NE;
    _Float16* WqT = p; p += WE;
    _Float16* WkT = p; p += WE;
    _Float16* WvT = p; p += WE;
    _Float16* WoT = p; p += WE;
    _Float16* Qf  = p; p += NE;
    _Float16* Kf  = p; p += NE;
    _Float16* Vt  = p; p += NE;
    _Float16* Oh  = xh;  // xh dead after Q projection -> reuse for attn output

    const int M = kSL * kBS;  // 8192

    // 1) down-convert activations + transpose-convert weights
    const int cvtB = 256;
    cvt_f16_kernel<<<dim3((unsigned)((NE + cvtB - 1) / cvtB)), dim3(cvtB), 0, stream>>>(x,  xh, (int)NE);
    cvt_f16_kernel<<<dim3((unsigned)((NE + cvtB - 1) / cvtB)), dim3(cvtB), 0, stream>>>(kx, kh, (int)NE);
    cvt_f16_kernel<<<dim3((unsigned)((NE + cvtB - 1) / cvtB)), dim3(cvtB), 0, stream>>>(vx, vh, (int)NE);
    transpose_w_kernel<<<dim3(32, 32), dim3(16, 16), 0, stream>>>(Wq, WqT);
    transpose_w_kernel<<<dim3(32, 32), dim3(16, 16), 0, stream>>>(Wk, WkT);
    transpose_w_kernel<<<dim3(32, 32), dim3(16, 16), 0, stream>>>(Wv, WvT);
    transpose_w_kernel<<<dim3(32, 32), dim3(16, 16), 0, stream>>>(Wo, WoT);

    // 2) projections (WMMA)
    dim3 gB(32, 8);
    dim3 gG(kD / 64, M / 256);  // (8, 32) — 32 rows x 64 cols per wave
    gemm512_wmma<<<gG, gB, 0, stream>>>(xh, WqT, bq, Qf, M, 0);
    gemm512_wmma<<<gG, gB, 0, stream>>>(kh, WkT, bk, Kf, M, 0);
    gemm512_wmma<<<gG, gB, 0, stream>>>(vh, WvT, bv, Vt, M, 1);

    // 3) causal flash attention (WMMA + async-to-LDS double buffering)
    attn_fa_wmma<<<dim3(kSL / 16, kBS * kH), dim3(32), 0, stream>>>(Qf, Kf, Vt, Oh);

    // 4) output projection -> f32 d_out (WMMA)
    gemm512_wmma<<<gG, gB, 0, stream>>>(Oh, WoT, bo, (void*)out, M, 2);
}
